// MultiHeadAttention_25262997636095
// MI455X (gfx1250) — matrix operand order repaired and checked by reading, not compiled, not run on hardware
//
#include <hip/hip_runtime.h>

// MI455X / gfx1250, wave32. bf16 WMMA (v_wmma_f32_16x16x32_bf16) everywhere.
typedef __attribute__((ext_vector_type(16))) __bf16 v16bf;
typedef __attribute__((ext_vector_type(8)))  float  v8f;
typedef __attribute__((ext_vector_type(4)))  int    v4i;

#define NEGF (-4294967295.0f)   // -2^32 + 1, reference causal fill

static __device__ __forceinline__ unsigned short f2bf(float f) {
  unsigned u = __builtin_bit_cast(unsigned, f);
  u = (u + 0x7FFFu + ((u >> 16) & 1u)) >> 16;   // round-to-nearest-even
  return (unsigned short)u;
}

union Frag { v16bf v; uint4 q[2]; unsigned short s[16]; };

// ---- async global->LDS copy (gfx1250 GLOBAL_LOAD_ASYNC_TO_LDS_B128) --------
#if __has_builtin(__builtin_amdgcn_global_load_async_to_lds_b128)
#define HAVE_ASYNC_LDS 1
static __device__ __forceinline__ void cp16_async(const unsigned short* gsrc,
                                                  unsigned short* ldst) {
  __builtin_amdgcn_global_load_async_to_lds_b128((v4i*)gsrc, (v4i*)ldst, 0, 0);
}
#else
#define HAVE_ASYNC_LDS 0
static __device__ __forceinline__ void cp16_async(const unsigned short* gsrc,
                                                  unsigned short* ldst) {
  *(uint4*)ldst = *(const uint4*)gsrc;
}
#endif
static __device__ __forceinline__ void async_drain() {
#if HAVE_ASYNC_LDS
  asm volatile("s_wait_asynccnt 0" ::: "memory");
#endif
}

// ---------------------------------------------------------------------------
// One-time weight convert+transpose: W f32 [k][n] -> Wt bf16 [n][k]
// ---------------------------------------------------------------------------
__global__ __launch_bounds__(256) void wcvt_k(const float* __restrict__ W,
                                              unsigned short* __restrict__ Wt) {
  __shared__ float T[32][33];
  const int n0 = blockIdx.x * 32, k0 = blockIdx.y * 32;
  const int j = threadIdx.x & 31, i0 = threadIdx.x >> 5;   // 8 rows/pass
  #pragma unroll
  for (int p = 0; p < 4; ++p) {
    const int i = i0 + p * 8;
    T[i][j] = W[(size_t)(k0 + i) * 1024 + n0 + j];
  }
  __syncthreads();
  #pragma unroll
  for (int p = 0; p < 4; ++p) {
    const int i = i0 + p * 8;
    Wt[(size_t)(n0 + i) * 1024 + k0 + j] = f2bf(T[j][i]);
  }
}

// ---------------------------------------------------------------------------
// GEMM: out[M=4096 x N=1024] = A[4096x1024] @ W[1024x1024] + bias
// Wt is pre-converted bf16 [n][k].
// MODE 0: store bf16 head-major  [h][b][s][qd]   (Q,K workspace)
// MODE 1: store bf16 transposed  [h][b][vd][s]   (V workspace)
// MODE 2: store f32 row-major    [m][n]          (final y)
// ABF16 : A is bf16 (heads) vs f32 (x)
// ---------------------------------------------------------------------------
template <int ABF16, int MODE>
__global__ __launch_bounds__(256) void gemm_k(const void* __restrict__ Aptr,
                                              const unsigned short* __restrict__ Wt,
                                              const float* __restrict__ bias,
                                              void* __restrict__ out) {
  __shared__ __align__(16) unsigned short As[128 * 40];  // [m][k] bf16, pad 8
  __shared__ __align__(16) unsigned short Bs[64 * 40];   // [n][k] bf16, pad 8

  const int tid  = threadIdx.x;
  const int lane = tid & 31;
  const int wave = tid >> 5;
  const int half = lane >> 4;
  const int l16  = lane & 15;
  const int wm   = wave & 3;        // 4 waves down M
  const int wn   = wave >> 2;       // 2 waves across N
  const int m0   = blockIdx.y * 128;
  const int n0   = blockIdx.x * 64;

  v8f c00 = {}, c01 = {}, c10 = {}, c11 = {};

  const float*          Af = (const float*)Aptr;
  const unsigned short* Ab = (const unsigned short*)Aptr;

  for (int k0 = 0; k0 < 1024; k0 += 32) {
    // ---- stage A tile 128x32 into LDS as bf16 ----
    if (ABF16) {
      int row = tid >> 2;                  // 0..63
      const int c8 = (tid & 3) * 8;        // 16B chunk
      #pragma unroll
      for (int p = 0; p < 2; ++p, row += 64)
        cp16_async(Ab + (size_t)(m0 + row) * 1024 + k0 + c8, &As[row * 40 + c8]);
    } else {
      int row = tid >> 3;                  // 0..31
      const int vc = (tid & 7) << 2;       // 0..28 step 4
      #pragma unroll
      for (int p = 0; p < 4; ++p, row += 32) {
        float4 d = *(const float4*)(Af + (size_t)(m0 + row) * 1024 + k0 + vc);
        unsigned short* dst = &As[row * 40 + vc];
        dst[0] = f2bf(d.x); dst[1] = f2bf(d.y);
        dst[2] = f2bf(d.z); dst[3] = f2bf(d.w);
      }
    }
    // ---- stage Wt tile 64x32 (bf16 [n][k]) into LDS ----
    {
      const int nn = tid >> 2;             // 0..63
      const int c8 = (tid & 3) * 8;
      cp16_async(Wt + (size_t)(n0 + nn) * 1024 + k0 + c8, &Bs[nn * 40 + c8]);
    }
    async_drain();
    __syncthreads();

    // ---- fragments (ISA 16-bit A 16x32 / B 32x16 layouts) ----
    Frag a0, a1, b0, b1;
    {
      const unsigned short* ap0 = &As[(wm * 32 + l16) * 40];
      const unsigned short* ap1 = &As[(wm * 32 + 16 + l16) * 40];
      a0.q[0] = *(const uint4*)(ap0 + 8 * half);
      a0.q[1] = *(const uint4*)(ap0 + 16 + 8 * half);
      a1.q[0] = *(const uint4*)(ap1 + 8 * half);
      a1.q[1] = *(const uint4*)(ap1 + 16 + 8 * half);
      const unsigned short* bp0 = &Bs[(wn * 32 + l16) * 40];
      const unsigned short* bp1 = &Bs[(wn * 32 + 16 + l16) * 40];
      b0.q[0] = *(const uint4*)(bp0 + 8 * half);
      b0.q[1] = *(const uint4*)(bp0 + 16 + 8 * half);
      b1.q[0] = *(const uint4*)(bp1 + 8 * half);
      b1.q[1] = *(const uint4*)(bp1 + 16 + 8 * half);
    }
    c00 = __builtin_amdgcn_wmma_f32_16x16x32_bf16(false, a0.v, false, b0.v, (short)0, c00, false, false);
    c01 = __builtin_amdgcn_wmma_f32_16x16x32_bf16(false, a0.v, false, b1.v, (short)0, c01, false, false);
    c10 = __builtin_amdgcn_wmma_f32_16x16x32_bf16(false, a1.v, false, b0.v, (short)0, c10, false, false);
    c11 = __builtin_amdgcn_wmma_f32_16x16x32_bf16(false, a1.v, false, b1.v, (short)0, c11, false, false);
    __syncthreads();
  }

  // ---- epilogue: bias + layout-specific store (compile-time MODE) ----
  v8f cc[2][2] = {{c00, c01}, {c10, c11}};
  #pragma unroll
  for (int i = 0; i < 2; ++i)
    #pragma unroll
    for (int j = 0; j < 2; ++j) {
      const int   n  = n0 + wn * 32 + j * 16 + l16;
      const float bv = bias[n];
      const int h = n >> 6, qd = n & 63;
      #pragma unroll
      for (int r = 0; r < 8; ++r) {
        const int   mg  = m0 + wm * 32 + i * 16 + r + 8 * half;
        const float val = cc[i][j][r] + bv;
        if (MODE == 2) {
          ((float*)out)[(size_t)mg * 1024 + n] = val;
        } else {
          const int bb = mg >> 10, ss = mg & 1023;
          unsigned short* ob = (unsigned short*)out;
          if (MODE == 0)
            ob[(((size_t)(h * 4 + bb) * 1024 + ss) << 6) + qd] = f2bf(val);
          else
            ob[((size_t)(h * 4 + bb) * 64 + qd) * 1024 + ss] = f2bf(val);
        }
      }
    }
}

// ---------------------------------------------------------------------------
// Attention: one wave = one 16-query tile of one (h,b). Flash-style online
// softmax; K/V fragments loaded direct from (L2-resident) bf16 workspace.
// raw_score (pre-mask, post-scale, FULL SxS) streamed nontemporally to d_out.
// ---------------------------------------------------------------------------
__global__ __launch_bounds__(128) void attn_k(const unsigned short* __restrict__ Q,
                                              const unsigned short* __restrict__ K,
                                              const unsigned short* __restrict__ Vt,
                                              const int* __restrict__ pmask,
                                              const int* __restrict__ stagep,
                                              float* __restrict__ raw,
                                              unsigned short* __restrict__ heads) {
  __shared__ __align__(16) unsigned short Pl[4][16 * 40];  // per-wave P tile

  const int tid  = threadIdx.x;
  const int lane = tid & 31;
  const int wave = tid >> 5;
  const int half = lane >> 4;
  const int l16  = lane & 15;
  const int gid  = blockIdx.x * 4 + wave;   // 0..4095
  const int hb   = gid >> 6;                // h*4 + b
  const int qt   = gid & 63;
  const int b    = hb & 3;
  const int q0   = qt * 16;
  const int causal = (stagep[0] == 1);
  unsigned short* Pw = &Pl[wave][0];

  // Q A-fragments (qd 0..31 and 32..63)
  Frag aQ0, aQ1;
  {
    const unsigned short* qp = Q + ((size_t)hb * 1024 + q0 + l16) * 64;
    aQ0.q[0] = *(const uint4*)(qp + 8 * half);
    aQ0.q[1] = *(const uint4*)(qp + 16 + 8 * half);
    aQ1.q[0] = *(const uint4*)(qp + 32 + 8 * half);
    aQ1.q[1] = *(const uint4*)(qp + 32 + 16 + 8 * half);
  }

  float mrun[8], lrun[8];
  #pragma unroll
  for (int r = 0; r < 8; ++r) { mrun[r] = -INFINITY; lrun[r] = 0.f; }
  v8f o0 = {}, o1 = {}, o2 = {}, o3 = {};

  const size_t rawbase = (size_t)hb * 1024 * 1024;

  for (int k0 = 0; k0 < 1024; k0 += 32) {        // 32 keys per chunk
    // prefetch next chunk's K rows and V columns (global_prefetch_b8)
    if (k0 + 32 < 1024) {
      __builtin_prefetch((const void*)(K + ((size_t)hb * 1024 + k0 + 32 + l16) * 64), 0, 0);
      __builtin_prefetch((const void*)(K + ((size_t)hb * 1024 + k0 + 48 + l16) * 64), 0, 0);
      __builtin_prefetch((const void*)(Vt + ((size_t)hb * 64 + l16) * 1024 + k0 + 32), 0, 0);
    }

    v8f sA = {}, sB = {};
    #pragma unroll
    for (int sub = 0; sub < 2; ++sub) {          // two 16-key B tiles
      const int kk = k0 + sub * 16;
      Frag bK0, bK1;
      const unsigned short* kp = K + ((size_t)hb * 1024 + kk + l16) * 64;
      bK0.q[0] = *(const uint4*)(kp + 8 * half);
      bK0.q[1] = *(const uint4*)(kp + 16 + 8 * half);
      bK1.q[0] = *(const uint4*)(kp + 32 + 8 * half);
      bK1.q[1] = *(const uint4*)(kp + 32 + 16 + 8 * half);
      v8f s = {};
      s = __builtin_amdgcn_wmma_f32_16x16x32_bf16(false, aQ0.v, false, bK0.v, (short)0, s, false, false);
      s = __builtin_amdgcn_wmma_f32_16x16x32_bf16(false, aQ1.v, false, bK1.v, (short)0, s, false, false);
      if (sub == 0) sA = s; else sB = s;
    }

    // scale, stream raw_score (pre-mask), apply causal + padding masks
    const int   colA = k0 + l16;
    const int   colB = k0 + 16 + l16;
    const float pmA  = (pmask[b * 1024 + colA] != 0) ? 1.0f : -INFINITY;
    const float pmB  = (pmask[b * 1024 + colB] != 0) ? 1.0f : -INFINITY;
    #pragma unroll
    for (int r = 0; r < 8; ++r) {
      const int row = q0 + r + 8 * half;
      float va = sA[r] * 0.125f;   // 1/sqrt(64)
      float vb = sB[r] * 0.125f;
      __builtin_nontemporal_store(va, raw + rawbase + (size_t)row * 1024 + colA);
      __builtin_nontemporal_store(vb, raw + rawbase + (size_t)row * 1024 + colB);
      if (causal) {
        if (colA > row) va = NEGF;
        if (colB > row) vb = NEGF;
      }
      sA[r] = va + pmA;
      sB[r] = vb + pmB;
    }

    // online softmax per row (rows live in 16-lane halves)
    #pragma unroll
    for (int r = 0; r < 8; ++r) {
      float v = fmaxf(sA[r], sB[r]);
      v = fmaxf(v, __shfl_xor(v, 1, 16));
      v = fmaxf(v, __shfl_xor(v, 2, 16));
      v = fmaxf(v, __shfl_xor(v, 4, 16));
      v = fmaxf(v, __shfl_xor(v, 8, 16));
      const float mnew  = fmaxf(mrun[r], v);
      const float alpha = __expf(mrun[r] - mnew);
      const float pa = __expf(sA[r] - mnew);
      const float pb = __expf(sB[r] - mnew);
      float sum = pa + pb;
      sum += __shfl_xor(sum, 1, 16);
      sum += __shfl_xor(sum, 2, 16);
      sum += __shfl_xor(sum, 4, 16);
      sum += __shfl_xor(sum, 8, 16);
      lrun[r] = lrun[r] * alpha + sum;
      mrun[r] = mnew;
      o0[r] *= alpha; o1[r] *= alpha; o2[r] *= alpha; o3[r] *= alpha;
      Pw[(r + 8 * half) * 40 + l16]      = f2bf(pa);   // P: C-layout -> LDS
      Pw[(r + 8 * half) * 40 + 16 + l16] = f2bf(pb);
    }
    asm volatile("s_wait_dscnt 0" ::: "memory");       // LDS store->load fence

    // reload P as 16x32 A-fragment
    Frag aP;
    {
      const unsigned short* pp = Pw + l16 * 40;
      aP.q[0] = *(const uint4*)(pp + 8 * half);
      aP.q[1] = *(const uint4*)(pp + 16 + 8 * half);
    }
    // O += P @ V   (4 B-tiles of vd)
    #pragma unroll
    for (int vt = 0; vt < 4; ++vt) {
      Frag bV;
      const unsigned short* vp = Vt + ((size_t)hb * 64 + vt * 16 + l16) * 1024 + k0;
      bV.q[0] = *(const uint4*)(vp + 8 * half);
      bV.q[1] = *(const uint4*)(vp + 16 + 8 * half);
      v8f& o = (vt == 0) ? o0 : (vt == 1) ? o1 : (vt == 2) ? o2 : o3;
      o = __builtin_amdgcn_wmma_f32_16x16x32_bf16(false, aP.v, false, bV.v, (short)0, o, false, false);
    }
  }

  // normalize and store heads (b,s, h*64+vd) bf16 for the output GEMM
  #pragma unroll
  for (int r = 0; r < 8; ++r) {
    const float inv = 1.0f / lrun[r];
    o0[r] *= inv; o1[r] *= inv; o2[r] *= inv; o3[r] *= inv;
  }
  const int h = hb >> 2;
  #pragma unroll
  for (int vt = 0; vt < 4; ++vt) {
    const v8f o = (vt == 0) ? o0 : (vt == 1) ? o1 : (vt == 2) ? o2 : o3;
    #pragma unroll
    for (int r = 0; r < 8; ++r) {
      const int srow = q0 + r + 8 * half;
      heads[(size_t)(b * 1024 + srow) * 1024 + h * 64 + vt * 16 + l16] = f2bf(o[r]);
    }
  }
}

// ---------------------------------------------------------------------------
extern "C" void kernel_launch(void* const* d_in, const int* in_sizes, int n_in,
                              void* d_out, int out_size, void* d_ws, size_t ws_size,
                              hipStream_t stream) {
  const float* x    = (const float*)d_in[0];
  const float* Wq   = (const float*)d_in[1];
  const float* bq   = (const float*)d_in[2];
  const float* Wk   = (const float*)d_in[3];
  const float* bk   = (const float*)d_in[4];
  const float* Wv   = (const float*)d_in[5];
  const float* bv   = (const float*)d_in[6];
  const float* Wo   = (const float*)d_in[7];
  const float* bo   = (const float*)d_in[8];
  const int*   pm   = (const int*)d_in[9];
  const int*   stg  = (const int*)d_in[10];

  float* y   = (float*)d_out;                       // (4,1024,1024) f32
  float* raw = y + (size_t)4 * 1024 * 1024;         // (64,1024,1024) f32

  const size_t MEG = (size_t)1024 * 1024;
  unsigned short* Qws = (unsigned short*)d_ws;      // 8 MB each
  unsigned short* Kws = Qws + 4 * MEG;
  unsigned short* Vtw = Kws + 4 * MEG;
  unsigned short* Hws = Vtw + 4 * MEG;
  unsigned short* Wqt = Hws + 4 * MEG;              // 2 MB each (bf16 [n][k])
  unsigned short* Wkt = Wqt + 1 * MEG;
  unsigned short* Wvt = Wkt + 1 * MEG;
  unsigned short* Wot = Wvt + 1 * MEG;

  dim3 gw(32, 32), gg(16, 32), bb(256);
  wcvt_k<<<gw, bb, 0, stream>>>(Wq, Wqt);
  wcvt_k<<<gw, bb, 0, stream>>>(Wk, Wkt);
  wcvt_k<<<gw, bb, 0, stream>>>(Wv, Wvt);
  wcvt_k<<<gw, bb, 0, stream>>>(Wo, Wot);

  gemm_k<0, 0><<<gg, bb, 0, stream>>>(x, Wqt, bq, Qws);   // Q [h,b,s,qd] bf16
  gemm_k<0, 0><<<gg, bb, 0, stream>>>(x, Wkt, bk, Kws);   // K [h,b,s,qd] bf16
  gemm_k<0, 1><<<gg, bb, 0, stream>>>(x, Wvt, bv, Vtw);   // V [h,b,vd,s] bf16

  attn_k<<<1024, 128, 0, stream>>>(Qws, Kws, Vtw, pm, stg, raw, Hws);

  gemm_k<1, 2><<<gg, bb, 0, stream>>>(Hws, Wot, bo, y);   // y = heads @ Wo + bo
}
